// DotProductAttention_44856638439986
// MI455X (gfx1250) — matrix operand order repaired and checked by reading, not compiled, not run on hardware
//
#include <hip/hip_runtime.h>
#include <hip/hip_bf16.h>

// -----------------------------------------------------------------------------
// Flash attention (masked, eval-mode) for gfx1250 / MI455X — transposed form,
// software-pipelined with double-buffered LDS staging.
//   queries/keys/values: (B=32, S=2048, D=64) fp32, valid_len: (B,) int32
//   out: (B, S, D) fp32
//
// Block = 8 wave32 = 128 query rows; each wave owns one 16-row q tile.
// Computes S^T = K.Q^T (M=keys, N=queries) and O^T = V^T.P^T (M=d, N=queries):
//   * softmax reduction over keys is IN-REGISTER plus one permlanex16 hop,
//   * m / l / corr / 1/l are one scalar per lane (query = lane),
//   * P^T B-fragment assembled in registers via an xor-16 swap trick,
//   * K / V^T tiles double-buffered in LDS: tile t+1's global loads are issued
//     BEFORE the compute on tile t and converted/stored after it -> one
//     barrier per tile and global latency hidden under WMMA work,
//   * padding mask hoisted out of the steady-state loop (last tile only),
//   * 8x v_wmma_f32_16x16x32_bf16 per 32-key iteration.
// -----------------------------------------------------------------------------

typedef __attribute__((ext_vector_type(16))) __bf16 v16bf;
typedef __attribute__((ext_vector_type(8)))  __bf16 v8bf;
typedef __attribute__((ext_vector_type(8)))  float  v8f;

namespace {
constexpr int kB     = 32;
constexpr int kS     = 2048;
constexpr int kD     = 64;
constexpr int kQT    = 16;   // query rows per wave
constexpr int kKT    = 32;   // keys per inner iteration (WMMA K for bf16)
constexpr int kWaves = 8;    // waves per block -> 128 query rows per block
constexpr int kKPitch = 72;  // K LDS row pitch (bf16 elems, padded)
constexpr int kVPitch = 40;  // V^T LDS row pitch
// softmax in base-2; scale (1/sqrt(64))*log2(e) folded into Q conversion
constexpr float kScaleLog2 = 0.125f * 1.44269504088896340736f;
constexpr float kNeg2      = -1442695.04f;   // -1e6 * log2(e)
}

__device__ __forceinline__ v8f wmma_bf16(v16bf a, v16bf b, v8f c) {
  return __builtin_amdgcn_wmma_f32_16x16x32_bf16(false, a, false, b,
                                                 (short)0, c, false, false);
}
__device__ __forceinline__ v16bf cat8(v8bf lo, v8bf hi) {
  return __builtin_shufflevector(lo, hi, 0, 1, 2, 3, 4, 5, 6, 7,
                                 8, 9, 10, 11, 12, 13, 14, 15);
}
// exchange with xor-16 partner lane (VALU permlane if available, else bpermute)
__device__ __forceinline__ float xor16(float x) {
#if __has_builtin(__builtin_amdgcn_permlanex16)
  int i = __builtin_bit_cast(int, x);
  i = __builtin_amdgcn_permlanex16(i, i, 0x76543210, 0xfedcba98, true, false);
  return __builtin_bit_cast(float, i);
#else
  return __shfl_xor(x, 16, 32);
#endif
}

__global__ __launch_bounds__(kWaves * 32)
void fa_fwd_kernel(const float* __restrict__ Q,
                   const float* __restrict__ K,
                   const float* __restrict__ V,
                   const int*   __restrict__ valid_len,
                   float*       __restrict__ O) {
  __shared__ __bf16 Kl[2][kKT][kKPitch];   // K tiles, row-major [key][d]
  __shared__ __bf16 Vt[2][kD][kVPitch];    // V tiles, transposed [d][key]

  const int lane = threadIdx.x & 31;
  const int wave = threadIdx.x >> 5;
  const int half = lane >> 4;       // 0: lanes 0-15, 1: lanes 16-31
  const int l16  = lane & 15;

  const int blocksPerBatch = kS / (kQT * kWaves);           // 16
  const int b  = blockIdx.x / blocksPerBatch;
  const int q0 = (blockIdx.x % blocksPerBatch) * (kQT * kWaves) + wave * kQT;

  const float* __restrict__ Qb = Q + ((size_t)b * kS + q0) * kD;
  const float* __restrict__ Kb = K + (size_t)b * kS * kD;
  const float* __restrict__ Vb = V + (size_t)b * kS * kD;
  const int vlen = valid_len[b];                 // uniform across the block
  const int ntiles = (vlen + kKT - 1) / kKT;     // >= 1; S % kKT == 0 -> no OOB

  // ---- Q B-fragments (pre-scaled): 32(d) x 16(query) per 32-d chunk --------
  v16bf qb0, qb1;
  {
    const float* qrow = Qb + (size_t)l16 * kD + 16 * half;
    #pragma unroll
    for (int e = 0; e < 16; ++e) {
      qb0[e] = (__bf16)(qrow[e]      * kScaleLog2);
      qb1[e] = (__bf16)(qrow[e + 32] * kScaleLog2);
    }
  }

  // ---- running state: one scalar per lane (query = q0 + l16) ---------------
  v8f ot0 = {}, ot1 = {}, ot2 = {}, ot3 = {};   // O^T accumulator (M=d, N=query)
  float mrun = -3.0e38f, lrun = 0.0f;

  // staging assignments (fixed per thread)
  const int krow = threadIdx.x >> 3;           // 0..31  key row for K staging
  const int kcol = (threadIdx.x & 7) * 8;      // 0..56  d column block
  const int vd   = threadIdx.x & 63;           // 0..63  d column for V staging
  const int vk   = (threadIdx.x >> 6) * 8;     // 0,8,16,24 key block

  float kx[8], vx[8];                          // in-flight staging registers
  auto stageLoad = [&](int tile) {
    const float4* ks = (const float4*)(Kb + (size_t)(tile * kKT + krow) * kD + kcol);
    *(float4*)&kx[0] = ks[0];
    *(float4*)&kx[4] = ks[1];
    #pragma unroll
    for (int i = 0; i < 8; ++i)
      vx[i] = Vb[(size_t)(tile * kKT + vk + i) * kD + vd];
    if (tile + 1 < ntiles)    // L2 prefetch one more tile ahead
      __builtin_prefetch(Kb + (size_t)((tile + 1) * kKT + krow) * kD + kcol, 0, 1);
  };
  auto stageStore = [&](int tile) {
    v8bf tk, tv;
    #pragma unroll
    for (int i = 0; i < 8; ++i) { tk[i] = (__bf16)kx[i]; tv[i] = (__bf16)vx[i]; }
    *(v8bf*)&Kl[tile & 1][krow][kcol] = tk;
    *(v8bf*)&Vt[tile & 1][vd][vk]     = tv;
  };

  // prologue: stage tile 0
  stageLoad(0);
  stageStore(0);

  for (int t = 0; t < ntiles; ++t) {
    __syncthreads();                 // buf[t&1] staged; buf[(t+1)&1] free
    const bool more = (t + 1 < ntiles);
    if (more) stageLoad(t + 1);      // global loads in flight during compute

    const __bf16 (*__restrict__ Klb)[kKPitch] = Kl[t & 1];
    const __bf16 (*__restrict__ Vtb)[kVPitch] = Vt[t & 1];
    const int n0 = t * kKT;

    // A-frag of K (M=key-in-tile, Kdim=d): two contiguous 16B chunks per lane
    auto loadKA = [&](int kt, int d0) -> v16bf {
      const __bf16* p = &Klb[kt * 16 + l16][d0 + 16 * half];
      return cat8(*(const v8bf*)p, *(const v8bf*)(p + 8));
    };

    // S^T: two 16(key) x 16(query) tiles
    v8f st0 = {}, st1 = {};
    st0 = wmma_bf16(loadKA(0,  0), qb0, st0);
    st0 = wmma_bf16(loadKA(0, 32), qb1, st0);
    st1 = wmma_bf16(loadKA(1,  0), qb0, st1);
    st1 = wmma_bf16(loadKA(1, 32), qb1, st1);

    // padding mask only on the final tile (block-uniform branch)
    if (t == ntiles - 1) {
      const int keybase = n0 + 8 * half;       // key = keybase + r (+16 for st1)
      #pragma unroll
      for (int r = 0; r < 8; ++r) {
        st0[r] = (keybase + r      >= vlen) ? kNeg2 : st0[r];
        st1[r] = (keybase + r + 16 >= vlen) ? kNeg2 : st1[r];
      }
    }

    // online softmax over keys: in-register reduce + one cross-half exchange
    float pm = fmaxf(st0[0], st1[0]);
    #pragma unroll
    for (int r = 1; r < 8; ++r) pm = fmaxf(pm, fmaxf(st0[r], st1[r]));
    pm = fmaxf(pm, xor16(pm));
    const float mnew = fmaxf(mrun, pm);
    const float corr = __builtin_amdgcn_exp2f(mrun - mnew);  // 0 on 1st iter

    float p0[8], p1[8], rs = 0.0f;
    #pragma unroll
    for (int r = 0; r < 8; ++r) {
      p0[r] = __builtin_amdgcn_exp2f(st0[r] - mnew);
      p1[r] = __builtin_amdgcn_exp2f(st1[r] - mnew);
      rs += p0[r] + p1[r];
    }
    rs += xor16(rs);
    lrun = lrun * corr + rs;
    mrun = mnew;
    ot0 *= corr; ot1 *= corr; ot2 *= corr; ot3 *= corr;

    // P^T B-frag (Kdim=key, N=query): elems e <-> key 16*half + e.
    v16bf pb;
    #pragma unroll
    for (int r = 0; r < 8; ++r) {
      const float send = half ? p0[r] : p1[r];
      const float recv = xor16(send);
      const float lo = half ? recv  : p0[r];   // key 16*half + r
      const float hi = half ? p1[r] : recv;    // key 16*half + 8 + r
      pb[r]     = (__bf16)lo;
      pb[r + 8] = (__bf16)hi;
    }

    // A-frag of V^T (M=d-in-tile, Kdim=key) straight from transposed LDS
    auto loadVA = [&](int dt) -> v16bf {
      const __bf16* p = &Vtb[dt * 16 + l16][16 * half];
      return cat8(*(const v8bf*)p, *(const v8bf*)(p + 8));
    };
    ot0 = wmma_bf16(loadVA(0), pb, ot0);
    ot1 = wmma_bf16(loadVA(1), pb, ot1);
    ot2 = wmma_bf16(loadVA(2), pb, ot2);
    ot3 = wmma_bf16(loadVA(3), pb, ot3);

    if (more) stageStore(t + 1);     // convert + LDS store after compute
  }

  // ---- epilogue: O^T C-layout -> row-major global, coalesced float4 x2 -----
  // element (d = 16*dt + r + 8*half, query = q0 + l16) sits in ot{dt}[r]
  const float inv = 1.0f / lrun;               // vlen >= 1 -> lrun > 0
  float* __restrict__ orow = O + ((size_t)b * kS + q0 + l16) * kD + 8 * half;
  const v8f o[4] = {ot0, ot1, ot2, ot3};
  #pragma unroll
  for (int dt = 0; dt < 4; ++dt) {
    float4 lo = make_float4(o[dt][0] * inv, o[dt][1] * inv,
                            o[dt][2] * inv, o[dt][3] * inv);
    float4 hi = make_float4(o[dt][4] * inv, o[dt][5] * inv,
                            o[dt][6] * inv, o[dt][7] * inv);
    *(float4*)(orow + 16 * dt)     = lo;
    *(float4*)(orow + 16 * dt + 4) = hi;
  }
}

extern "C" void kernel_launch(void* const* d_in, const int* in_sizes, int n_in,
                              void* d_out, int out_size, void* d_ws, size_t ws_size,
                              hipStream_t stream) {
  const float* Q  = (const float*)d_in[0];
  const float* K  = (const float*)d_in[1];
  const float* V  = (const float*)d_in[2];
  const int*   vl = (const int*)d_in[3];
  float* out = (float*)d_out;

  const int blocksPerBatch = kS / (kQT * kWaves);   // 16
  dim3 grid(kB * blocksPerBatch);                   // 512 blocks
  dim3 block(kWaves * 32);                          // 256 threads = 8 wave32
  fa_fwd_kernel<<<grid, block, 0, stream>>>(Q, K, V, vl, out);
}
